// LightweightDecoder_62491774157365
// MI455X (gfx1250) — matrix operand order repaired and checked by reading, not compiled, not run on hardware
//
#include <hip/hip_runtime.h>
#include <hip/hip_bf16.h>
#include <math.h>
#include <stdint.h>

// ---------------------------------------------------------------------------
// Tacotron-style teacher-forced decoder for MI455X (gfx1250, wave32, WMMA).
// All GEMMs use v_wmma_f32_16x16x32_f16 (f16 inputs, f32 accumulate); GEMM
// W tiles are staged with global_load_async_to_lds_b128 (ASYNCcnt).
// Sizes: B=32, S=512, T=500, E=M=D=256, H=4, HD=64.
// ---------------------------------------------------------------------------

typedef __attribute__((ext_vector_type(16))) _Float16 v16h;
typedef __attribute__((ext_vector_type(8)))  _Float16 v8h;
typedef __attribute__((ext_vector_type(4)))  _Float16 v4h;
typedef __attribute__((ext_vector_type(8)))  float    v8f;
typedef __attribute__((ext_vector_type(4)))  float    v4f;
typedef __attribute__((ext_vector_type(16))) float    v16f;

#define DEV __device__ __forceinline__

DEV float sigmoid_fast(float x) {
  return __builtin_amdgcn_rcpf(1.0f + __expf(-x));      // v_exp + v_rcp
}
DEV float tanh_fast(float x) {
  float xc = fminf(fmaxf(x, -15.f), 15.f);
  float e = __expf(2.0f * xc);
  return (e - 1.0f) * __builtin_amdgcn_rcpf(e + 1.0f);
}

DEV v8f wmma_f16(v16h a, v16h b, v8f c) {
  return __builtin_amdgcn_wmma_f32_16x16x32_f16(false, a, false, b,
                                                (short)0, c, false, false);
}
DEV v16h cat8(v8h lo, v8h hi) {
  return __builtin_shufflevector(lo, hi, 0, 1, 2, 3, 4, 5, 6, 7,
                                 8, 9, 10, 11, 12, 13, 14, 15);
}
// issue a per-lane 16B global->LDS async copy (tracked by ASYNCcnt)
DEV void async_copy_b128(const float* gsrc, float* ldst) {
  unsigned lds = (unsigned)(uintptr_t)ldst;  // low 32 bits = LDS byte offset
  asm volatile("global_load_async_to_lds_b128 %0, %1, off"
               :: "v"(lds), "v"(gsrc) : "memory");
}
DEV void async_wait0() { asm volatile("s_wait_asynccnt 0" ::: "memory"); }

enum { GF_RELU = 2, GF_ADDSRC = 4, GF_TB2BT = 8, GF_GATES = 16, GF_VT = 32 };

// ---------------------------------------------------------------------------
// build X = teacher-forced decoder inputs, rows in TB order (r = t*32 + b).
// Also copy into XCat[:, 256:512].
// ---------------------------------------------------------------------------
__global__ void build_x_kernel(const float* __restrict__ mel_t,
                               float* __restrict__ X, float* __restrict__ XCat) {
  int idx = (blockIdx.x * blockDim.x + threadIdx.x) * 4;
  if (idx >= 16000 * 256) return;
  int r = idx >> 8, k = idx & 255;
  int t = r >> 5, b = r & 31;
  v4f v = {};
  if (t > 0) v = *(const v4f*)&mel_t[((size_t)b * 500 + (t - 1)) * 256 + k];
  *(v4f*)&X[(size_t)r * 256 + k] = v;
  *(v4f*)&XCat[(size_t)r * 512 + 256 + k] = v;
}

// one-time w_hh f32 -> f16 (row-major [1024][256]) for the scan's B fragments
__global__ void cvt_whh_kernel(const float* __restrict__ w,
                               _Float16* __restrict__ o) {
  int idx = (blockIdx.x * blockDim.x + threadIdx.x) * 4;
  if (idx >= 1024 * 256) return;
  v4f v = *(const v4f*)&w[idx];
  *(v4h*)&o[idx] = __builtin_convertvector(v, v4h);
}

// one-time conv weight pack: [o][i][dt] -> [dt][o][i] (dense rows for GEMM)
__global__ void pack_convw_kernel(const float* __restrict__ w,
                                  float* __restrict__ o) {
  int idx = blockIdx.x * blockDim.x + threadIdx.x;
  if (idx >= 3 * 256 * 256) return;
  int dt = idx >> 16, rem = idx & 65535;
  int n = rem >> 8, k = rem & 255;
  o[idx] = w[(size_t)n * 768 + k * 3 + dt];
}

// one-time bias fold: LSTM total gate bias = b_ih + b_hh
__global__ void bias_sum_kernel(const float* __restrict__ a,
                                const float* __restrict__ b,
                                float* __restrict__ o) {
  int i = blockIdx.x * blockDim.x + threadIdx.x;
  if (i < 1024) o[i] = a[i] + b[i];
}

// ---------------------------------------------------------------------------
// Generic WMMA GEMM: C[M x N] = sum_tap A_shift(tap)[M x K] @ W_tap^T (+bias).
//   NTAPS=1: plain GEMM. NTAPS=3: fused conv1d(k=3) — A rows shifted by
//   tShift+tap within Tg groups (zero pad), W advanced by wTap per tap.
//   GF_TB2BT: output row m (= t*32+b) remapped to b*Tbt + t.
//   GF_GATES: output in scan-native layout [t][1024][32] (vectorized stores).
//   GF_VT:    output V transposed [b][h][d][512] (vectorized stores).
// Block: 256 threads (8 waves); tile 64(M) x 128(N); each wave computes four
// 16x16 tiles sharing one B fragment. W tiles staged raw via async-to-LDS
// (each Ws element is consumed by exactly one wave -> cvt in fragment build).
// ---------------------------------------------------------------------------
template <int FLAGS, int NTAPS>
__global__ __launch_bounds__(256)
void gemm_wmma_kernel(const float* __restrict__ A, const float* __restrict__ W,
                      const float* __restrict__ bias, float* __restrict__ C,
                      const float* __restrict__ addsrc,
                      int M, int N, int K, int lda, int ldc, int cColOff,
                      int wRow, int wTap, int tShift, int Tg, int Tbt) {
  __shared__ __align__(64) _Float16 As[64 * 32];
  __shared__ __align__(64) float    Wsf[128 * 32];
  const int tid = threadIdx.x;
  const int lane = tid & 31, wave = tid >> 5;
  const int half = lane >> 4, lidx = lane & 15;
  const int mBase = blockIdx.y * 64;
  const int nBlock = blockIdx.x * 128;
  v8f acc[4] = {};
#pragma unroll
  for (int tap = 0; tap < NTAPS; ++tap) {
    const float* Wt = W + (size_t)tap * wTap;
    const int shift = tShift + tap;
    for (int kb = 0; kb < K; kb += 32) {
      {  // stage A tile (64x32): 8 consecutive floats per thread, packed cvt
        int idx = tid * 8;
        int m = idx >> 5, k = idx & 31;
        int row = mBase + m;
        const float* src = nullptr;
        if (Tg > 0) {
          int bb = row / Tg, tt = row - bb * Tg;
          int st = tt + shift;
          if (st >= 0 && st < Tg) src = &A[(size_t)(bb * Tg + st) * lda + kb + k];
        } else {
          src = &A[(size_t)row * lda + kb + k];
        }
        v4f v0 = {}, v1 = {};
        if (src) { v0 = *(const v4f*)src; v1 = *(const v4f*)(src + 4); }
        *(v4h*)&As[idx]     = __builtin_convertvector(v0, v4h);
        *(v4h*)&As[idx + 4] = __builtin_convertvector(v1, v4h);
      }
#pragma unroll
      for (int i = 0; i < 4; ++i) {  // stage W tile (128x32) raw: async b128
        int idx = tid * 4 + i * 1024;
        int n = idx >> 5, k = idx & 31;
        async_copy_b128(&Wt[(size_t)(nBlock + n) * wRow + kb + k], &Wsf[idx]);
      }
      async_wait0();
      __syncthreads();
      // B fragment: 16 contiguous f32 in LDS -> packed cvt (one wave per row)
      v16f wv = *(const v16f*)&Wsf[(wave * 16 + lidx) * 32 + half * 16];
      v16h bf = __builtin_convertvector(wv, v16h);
#pragma unroll
      for (int mt = 0; mt < 4; ++mt) {
        v16h af = *(const v16h*)&As[(mt * 16 + lidx) * 32 + half * 16];
        acc[mt] = wmma_f16(af, bf, acc[mt]);
      }
      __syncthreads();
    }
  }
  const int n = nBlock + wave * 16 + lidx;
  const float bv = bias ? bias[n] : 0.f;
#pragma unroll
  for (int mt = 0; mt < 4; ++mt) {
    if (FLAGS == GF_GATES) {
      // 8 consecutive m-rows -> contiguous along the b dimension of [t][n][32]
      int m0 = mBase + mt * 16 + 8 * half;
      size_t base = ((size_t)(m0 >> 5) * 1024 + n) * 32 + (m0 & 31);
      v4f lo = {acc[mt][0] + bv, acc[mt][1] + bv, acc[mt][2] + bv, acc[mt][3] + bv};
      v4f hi = {acc[mt][4] + bv, acc[mt][5] + bv, acc[mt][6] + bv, acc[mt][7] + bv};
      *(v4f*)&C[base] = lo;
      *(v4f*)&C[base + 4] = hi;
    } else if (FLAGS == GF_VT) {
      // rows are b*512+s -> contiguous along s of VT[b][h][d][512]
      int m0 = mBase + mt * 16 + 8 * half;
      size_t base = (((size_t)(m0 >> 9) * 4 + (n >> 6)) * 64 + (n & 63)) * 512 +
                    (m0 & 511);
      v4f lo = {acc[mt][0] + bv, acc[mt][1] + bv, acc[mt][2] + bv, acc[mt][3] + bv};
      v4f hi = {acc[mt][4] + bv, acc[mt][5] + bv, acc[mt][6] + bv, acc[mt][7] + bv};
      *(v4f*)&C[base] = lo;
      *(v4f*)&C[base + 4] = hi;
    } else {
#pragma unroll
      for (int r = 0; r < 8; ++r) {
        int m = mBase + mt * 16 + r + 8 * half;
        size_t ci;
        if (FLAGS & GF_TB2BT) {
          int bb = m & 31, tt = m >> 5;
          ci = (size_t)(bb * Tbt + tt) * ldc + cColOff + n;
        } else {
          ci = (size_t)m * ldc + cColOff + n;
        }
        float v = acc[mt][r] + bv;
        if (FLAGS & GF_RELU)   v = fmaxf(v, 0.f);
        if (FLAGS & GF_ADDSRC) v += addsrc[ci];
        C[ci] = v;
      }
    }
  }
}

template <int FLAGS, int NTAPS>
static void launch_gemm(hipStream_t stream, const float* A, const float* W,
                        const float* bias, float* C, const float* addsrc,
                        int M, int N, int K, int lda, int ldc, int cColOff,
                        int wRow, int wTap, int tShift, int Tg, int Tbt) {
  dim3 g(N / 128, M / 64), blk(256);
  gemm_wmma_kernel<FLAGS, NTAPS><<<g, blk, 0, stream>>>(
      A, W, bias, C, addsrc, M, N, K, lda, ldc, cColOff, wRow, wTap,
      tShift, Tg, Tbt);
}

// ---------------------------------------------------------------------------
// Fused attention for a 16-row t-block of one (b, h):
//   S = (Q*scale) Kᵀ  (WMMA) -> softmax over S=512 (LDS) -> ctx = P V (WMMA).
// One wave per block. Q rows in TB order; K rows b*S+s; V pre-transposed.
// ---------------------------------------------------------------------------
__global__ __launch_bounds__(32)
void attn_kernel(const float* __restrict__ Q, const float* __restrict__ Kb,
                 const float* __restrict__ VT, float* __restrict__ Ctx) {
  __shared__ __align__(64) _Float16 Qs[16 * 64];
  __shared__ __align__(64) float    Sc[16 * 512];
  __shared__ __align__(64) _Float16 Ps[16 * 512];
  __shared__ float red[32];
  const int b = blockIdx.z, h = blockIdx.y, tblk = blockIdx.x;
  const int lane = threadIdx.x;
  const int half = lane >> 4, lidx = lane & 15;
  for (int idx = lane; idx < 16 * 64; idx += 32) {
    int i = idx >> 6, d = idx & 63;
    int t = tblk * 16 + i; if (t > 499) t = 499;           // clamp tail rows
    Qs[idx] = (_Float16)(Q[(size_t)(t * 32 + b) * 256 + h * 64 + d] * 0.125f);
  }
  __syncthreads();
  for (int sb = 0; sb < 32; ++sb) {                         // S in 16-col tiles
    v8f acc = {};
#pragma unroll
    for (int kc = 0; kc < 2; ++kc) {                        // HD=64 -> 2 k-steps
      v16h af = *(const v16h*)&Qs[lidx * 64 + kc * 32 + half * 16];
      v16f kv = *(const v16f*)&Kb[(size_t)(b * 512 + sb * 16 + lidx) * 256 +
                                  h * 64 + kc * 32 + half * 16];
      acc = wmma_f16(af, __builtin_convertvector(kv, v16h), acc);
    }
#pragma unroll
    for (int r = 0; r < 8; ++r)
      Sc[(r + 8 * half) * 512 + sb * 16 + lidx] = acc[r];
  }
  __syncthreads();
  // softmax: lane (half,lidx) handles row lidx, half of the 512 columns
  const int rowOff = lidx * 512 + half * 256;
  float mx = -1e30f;
  for (int c = 0; c < 256; c += 4) {
    v4f v = *(const v4f*)&Sc[rowOff + c];
    mx = fmaxf(fmaxf(fmaxf(mx, v.x), v.y), fmaxf(v.z, v.w));
  }
  red[lane] = mx;
  __syncthreads();
  mx = fmaxf(red[lidx], red[16 + lidx]);
  float sum = 0.f;
  for (int c = 0; c < 256; c += 4) {
    v4f v = *(const v4f*)&Sc[rowOff + c];
    v4f e = {__expf(v.x - mx), __expf(v.y - mx), __expf(v.z - mx), __expf(v.w - mx)};
    sum += e.x + e.y + e.z + e.w;
    *(v4f*)&Sc[rowOff + c] = e;
  }
  __syncthreads();
  red[lane] = sum;
  __syncthreads();
  float inv = __builtin_amdgcn_rcpf(red[lidx] + red[16 + lidx]);
  for (int c = 0; c < 256; c += 4) {
    v4f v = *(const v4f*)&Sc[rowOff + c] * inv;
    *(v4h*)&Ps[rowOff + c] = __builtin_convertvector(v, v4h);
  }
  __syncthreads();
  // ctx = P (16x512) @ V (512x64): 4 n-tiles x 16 k-steps; VT rows contiguous
#pragma unroll
  for (int nt = 0; nt < 4; ++nt) {
    const float* vrow = &VT[(((size_t)b * 4 + h) * 64 + nt * 16 + lidx) * 512];
    v8f acc = {};
    for (int kc = 0; kc < 16; ++kc) {
      v16h af = *(const v16h*)&Ps[lidx * 512 + kc * 32 + half * 16];
      v16f vv = *(const v16f*)&vrow[kc * 32 + half * 16];
      acc = wmma_f16(af, __builtin_convertvector(vv, v16h), acc);
    }
#pragma unroll
    for (int r = 0; r < 8; ++r) {
      int m = r + 8 * half;
      int t = tblk * 16 + m;
      if (t < 500)
        Ctx[(size_t)(t * 32 + b) * 256 + h * 64 + nt * 16 + lidx] = acc[r];
    }
  }
}

// ---------------------------------------------------------------------------
// Sequential LSTM scan: 1 block, 16 waves, ping-pong h in LDS (1 barrier/step).
// Per step: gates = G2[t] + h@Whhᵀ (WMMA), fast transcendentals, c in regs.
// G2 layout [t][1024][32] (bias b_ih+b_hh already folded in by the GEMM);
// Whh16 fragments are single contiguous 32B loads; next step's G2 slice is
// prefetched (global_prefetch_b8) during the matmul phase.
// ---------------------------------------------------------------------------
__global__ __launch_bounds__(512)
void lstm_scan_kernel(const float* __restrict__ G2,
                      const _Float16* __restrict__ Whh16,
                      float* __restrict__ Hall) {
  __shared__ __align__(64) _Float16 Hs[2][32 * 256];
  const int tid = threadIdx.x;
  const int lane = tid & 31, wave = tid >> 5;
  const int half = lane >> 4, lidx = lane & 15;
  const int ncol = wave * 16 + lidx;
  for (int idx = tid; idx < 32 * 256; idx += 512) Hs[0][idx] = (_Float16)0.f;
  float c_reg[2][8] = {};
  __syncthreads();
  int cur = 0;
  for (int t = 0; t < 500; ++t) {
    if (t + 1 < 500)  // hide L2 latency of next step's gate slice
      __builtin_prefetch(&G2[(size_t)(t + 1) * 32768 + (size_t)tid * 64], 0, 1);
    v8f acc[2][4] = {};
    for (int kb = 0; kb < 8; ++kb) {                        // K = 256
      v16h af[2];
#pragma unroll
      for (int mt = 0; mt < 2; ++mt)
        af[mt] = *(const v16h*)&Hs[cur][(mt * 16 + lidx) * 256 + kb * 32 + half * 16];
#pragma unroll
      for (int g = 0; g < 4; ++g) {
        v16h bf = *(const v16h*)&Whh16[(size_t)(g * 256 + ncol) * 256 +
                                       kb * 32 + half * 16];
        acc[0][g] = wmma_f16(af[0], bf, acc[0][g]);
        acc[1][g] = wmma_f16(af[1], bf, acc[1][g]);
      }
    }
#pragma unroll
    for (int mt = 0; mt < 2; ++mt) {
      int m0 = mt * 16 + 8 * half;                          // 8 consecutive b's
      float gv[4][8];
#pragma unroll
      for (int g = 0; g < 4; ++g) {
        size_t off = ((size_t)t * 1024 + g * 256 + ncol) * 32 + m0;
        v4f lo = *(const v4f*)&G2[off];
        v4f hi = *(const v4f*)&G2[off + 4];
        gv[g][0] = lo.x; gv[g][1] = lo.y; gv[g][2] = lo.z; gv[g][3] = lo.w;
        gv[g][4] = hi.x; gv[g][5] = hi.y; gv[g][6] = hi.z; gv[g][7] = hi.w;
      }
#pragma unroll
      for (int r = 0; r < 8; ++r) {
        float gi = acc[mt][0][r] + gv[0][r];
        float gf = acc[mt][1][r] + gv[1][r];
        float gg = acc[mt][2][r] + gv[2][r];
        float go = acc[mt][3][r] + gv[3][r];
        float cc = sigmoid_fast(gf) * c_reg[mt][r] + sigmoid_fast(gi) * tanh_fast(gg);
        c_reg[mt][r] = cc;
        float hv = sigmoid_fast(go) * tanh_fast(cc);
        Hall[((size_t)t * 32 + m0 + r) * 256 + ncol] = hv;
        Hs[cur ^ 1][(m0 + r) * 256 + ncol] = (_Float16)hv;  // other buffer
      }
    }
    __syncthreads();  // writes to Hs[cur^1] visible before next step's reads
    cur ^= 1;
  }
}

// stop = Hall @ stop_wᵀ + stop_b (N=1): plain VALU kernel, TB -> BT remap.
__global__ void stop_kernel(const float* __restrict__ Hall,
                            const float* __restrict__ sw,
                            const float* __restrict__ sb,
                            float* __restrict__ out) {
  int r = blockIdx.x * blockDim.x + threadIdx.x;
  if (r >= 16000) return;
  float s = sb[0];
  for (int k = 0; k < 256; k += 4) {
    v4f hv = *(const v4f*)&Hall[(size_t)r * 256 + k];
    v4f wv = *(const v4f*)&sw[k];
    s += hv.x * wv.x + hv.y * wv.y + hv.z * wv.z + hv.w * wv.w;
  }
  int b = r & 31, t = r >> 5;
  out[b * 500 + t] = s;
}

// ---------------------------------------------------------------------------
extern "C" void kernel_launch(void* const* d_in, const int* in_sizes, int n_in,
                              void* d_out, int out_size, void* d_ws, size_t ws_size,
                              hipStream_t stream) {
  (void)in_sizes; (void)n_in; (void)out_size; (void)ws_size;
  const float* enc        = (const float*)d_in[0];
  const float* mel_t      = (const float*)d_in[1];
  const float* in_proj_w  = (const float*)d_in[2];
  const float* in_proj_b  = (const float*)d_in[3];
  const float* out_proj_w = (const float*)d_in[4];
  const float* out_proj_b = (const float*)d_in[5];
  const float* w_ih       = (const float*)d_in[6];
  const float* w_hh       = (const float*)d_in[7];
  const float* b_ih       = (const float*)d_in[8];
  const float* b_hh       = (const float*)d_in[9];
  const float* mel_w      = (const float*)d_in[10];
  const float* mel_b      = (const float*)d_in[11];
  const float* stop_w     = (const float*)d_in[12];
  const float* stop_b     = (const float*)d_in[13];
  const float* c1w        = (const float*)d_in[14];
  const float* c1b        = (const float*)d_in[15];
  const float* c2w        = (const float*)d_in[16];
  const float* c2b        = (const float*)d_in[17];
  float* out = (float*)d_out;
  float* ws  = (float*)d_ws;

  // workspace layout (floats); Gbuf aliases the dead K/VT/Q/ctx region,
  // conv scratch C1 aliases the dead XCat region. All stream-ordered.
  float* Kbuf = ws + 0;         // 4,194,304
  float* VT   = ws + 4194304;   // 4,194,304  V transposed [b][h][64][512]
  float* Qbuf = ws + 8388608;   // 4,096,000
  float* Ctx  = ws + 12484608;  // 4,096,000  (region A ends @ 16,580,608)
  float* Gbuf = ws;             // 16,384,000 (alias region A) [t][1024][32]
  float* X    = ws + 16580608;  // 4,096,000
  float* Hall = ws + 20676608;  // 4,096,000
  float* XCat = ws + 24772608;  // 8,192,000
  float* C1   = XCat;           // conv1 scratch, XCat dead after gates GEMM
  _Float16* Whh16 = (_Float16*)(ws + 32964608);  // 262,144 halves
  float* c1wP = ws + 33095680;  // 196,608 packed [dt][o][i]
  float* c2wP = ws + 33292288;  // 196,608
  float* bsum = ws + 33488896;  // 1,024 (b_ih + b_hh), total ~134 MB

  float* melOut  = out;            // [B,T,256]
  float* melPost = out + 4096000;  // [B,T,256]
  float* stops   = out + 8192000;  // [B,T,1]

  // 0) one-time weight transforms + teacher-forced inputs
  cvt_whh_kernel<<<256, 256, 0, stream>>>(w_hh, Whh16);
  pack_convw_kernel<<<768, 256, 0, stream>>>(c1w, c1wP);
  pack_convw_kernel<<<768, 256, 0, stream>>>(c2w, c2wP);
  bias_sum_kernel<<<4, 256, 0, stream>>>(b_ih, b_hh, bsum);
  build_x_kernel<<<4000, 256, 0, stream>>>(mel_t, X, XCat);
  // 1-2) K / V projections (V written transposed for contiguous PV fragments)
  launch_gemm<0, 1>(stream, enc, in_proj_w + 256 * 256, in_proj_b + 256, Kbuf,
                    nullptr, 16384, 256, 256, 256, 256, 0, 256, 0, 0, 0, 0);
  launch_gemm<GF_VT, 1>(stream, enc, in_proj_w + 512 * 256, in_proj_b + 512, VT,
                        nullptr, 16384, 256, 256, 256, 256, 0, 256, 0, 0, 0, 0);
  // 3) Q for all steps
  launch_gemm<0, 1>(stream, X, in_proj_w, in_proj_b, Qbuf,
                    nullptr, 16000, 256, 256, 256, 256, 0, 256, 0, 0, 0, 0);
  // 4) fused attention for all steps
  attn_kernel<<<dim3(32, 4, 32), dim3(32), 0, stream>>>(Qbuf, Kbuf, VT, Ctx);
  // 5) attn output projection, written directly into XCat[:, :256]
  launch_gemm<0, 1>(stream, Ctx, out_proj_w, out_proj_b, XCat,
                    nullptr, 16000, 256, 256, 256, 512, 0, 256, 0, 0, 0, 0);
  // 6) input-gate precompute in scan-native layout, bias = b_ih + b_hh
  launch_gemm<GF_GATES, 1>(stream, XCat, w_ih, bsum, Gbuf,
                           nullptr, 16000, 1024, 512, 512, 1024, 0, 512, 0, 0, 0, 0);
  // 7) sequential LSTM scan (the only latency-bound part)
  lstm_scan_kernel<<<1, 512, 0, stream>>>(Gbuf, Whh16, Hall);
  // 8) mel projection, TB->BT output remap
  launch_gemm<GF_TB2BT, 1>(stream, Hall, mel_w, mel_b, melOut,
                           nullptr, 16000, 256, 256, 256, 256, 0, 256, 0, 0, 0, 500);
  // 9) stop projection
  stop_kernel<<<(16000 + 255) / 256, 256, 0, stream>>>(Hall, stop_w, stop_b, stops);
  // 10) postnet conv1: single fused 3-tap GEMM + bias + ReLU
  launch_gemm<GF_RELU, 3>(stream, melOut, c1wP, c1b, C1, nullptr,
                          16000, 256, 256, 256, 256, 0, 256, 65536, -1, 500, 0);
  // 11) postnet conv2: single fused 3-tap GEMM + bias + residual mel add
  launch_gemm<GF_ADDSRC, 3>(stream, C1, c2wP, c2b, melPost, melOut,
                            16000, 256, 256, 256, 256, 0, 256, 65536, -1, 500, 0);
}
